// AttentionGeneMLP_21328807592185
// MI455X (gfx1250) — matrix operand order repaired and checked by reading, not compiled, not run on hardware
//
#include <hip/hip_runtime.h>
#include <hip/hip_bf16.h>

typedef __attribute__((ext_vector_type(16))) _Float16 v16h;
typedef __attribute__((ext_vector_type(8)))  _Float16 v8h;
typedef __attribute__((ext_vector_type(8)))  float    v8f;

#define B_   128
#define S_   40000
#define G_   4000
#define E_   16
#define NI_  4
#define H1_  1024
#define H2_  256
#define EPS_ 1e-5f

__device__ __forceinline__ float gelu_exact(float t) {
  // jax.nn.gelu(approximate=False): 0.5*t*(1+erf(t/sqrt(2)))
  return 0.5f * t * (1.0f + erff(t * 0.70710678118654752f));
}

// ------------------------------------------------------------------
// K1: NI=4 distinct impact embeddings -> 4 (scale, bias) table entries
// ------------------------------------------------------------------
__global__ void impact_tables_k(const float* __restrict__ emb, const float* __restrict__ proj_w,
                                const float* __restrict__ proj_b, const float* __restrict__ ln_i_w,
                                const float* __restrict__ ln_i_b, const float* __restrict__ scale_w,
                                const float* __restrict__ scale_b, const float* __restrict__ bias_w,
                                const float* __restrict__ bias_b, float* __restrict__ tab) {
  int ni = threadIdx.x;
  if (ni >= NI_) return;
  float h[E_], h2[E_];
  #pragma unroll
  for (int e = 0; e < E_; ++e) h[e] = emb[ni * E_ + e];
  #pragma unroll
  for (int e = 0; e < E_; ++e) {
    float s = proj_b[e];
    #pragma unroll
    for (int j = 0; j < E_; ++j) s += proj_w[e * E_ + j] * h[j];
    h2[e] = s;
  }
  float mean = 0.f;
  #pragma unroll
  for (int e = 0; e < E_; ++e) mean += h2[e];
  mean *= (1.0f / E_);
  float var = 0.f;
  #pragma unroll
  for (int e = 0; e < E_; ++e) { float d = h2[e] - mean; var += d * d; }
  var *= (1.0f / E_);
  float inv = rsqrtf(var + EPS_);
  float sc = scale_b[0], bi = bias_b[0];
  #pragma unroll
  for (int e = 0; e < E_; ++e) {
    float t = (h2[e] - mean) * inv * ln_i_w[e] + ln_i_b[e];
    t = gelu_exact(t);
    sc += t * scale_w[e];
    bi += t * bias_w[e];
  }
  tab[ni] = sc;
  tab[NI_ + ni] = bi;
}

// ------------------------------------------------------------------
// K2: xs = x * 2*sigmoid(x*scale[idx]+bias[idx]); write transposed xs_t[S][B]
// 32x32 LDS tile transpose, S and B are exact multiples of 32.
// ------------------------------------------------------------------
__global__ void attn_transpose_k(const float* __restrict__ x, const int* __restrict__ iidx,
                                 const float* __restrict__ tab, float* __restrict__ xs_t) {
  __shared__ float tile[32][33];
  int sBase = blockIdx.x * 32, bBase = blockIdx.y * 32;
  int tx = threadIdx.x, ty = threadIdx.y;
  int s = sBase + tx, b = bBase + ty;
  int ii = iidx[s];
  float sc = tab[ii], bi = tab[NI_ + ii];
  float xv = x[(size_t)b * S_ + s];
  float t = xv * sc + bi;
  float sig = 1.0f / (1.0f + expf(-t));
  tile[ty][tx] = xv * 2.0f * sig;
  __syncthreads();
  int so = sBase + ty, bo = bBase + tx;
  xs_t[(size_t)so * B_ + bo] = tile[tx][ty];
}

// ------------------------------------------------------------------
// K3: masked linear SNP->gene, exploiting 1-nonzero-per-column structure.
// One block per gene row; stream mask row (float4), ballot-compact nonzeros
// (deterministic: no atomics), gather mw and xs_t only at nonzeros.
// ------------------------------------------------------------------
#define GCHUNK 512
__global__ void gene_accum_k(const float* __restrict__ mask, const float* __restrict__ mw,
                             const float* __restrict__ xs_t, const float* __restrict__ mb,
                             float* __restrict__ gpre_t) {
  int g = blockIdx.x;
  int tid = threadIdx.x;                 // 128 threads = 4 waves (wave32)
  int lane = tid & 31, wid = tid >> 5;
  const size_t row = (size_t)g * S_;
  __shared__ float wv[GCHUNK];
  __shared__ unsigned bm[4][4];          // [wave][sub-element]
  float acc = 0.f;
  for (int base = 0; base < S_; base += GCHUNK) {
    int s0 = base + tid * 4;
    float4 m4 = make_float4(0.f, 0.f, 0.f, 0.f);
    if (s0 + 3 < S_) {
      m4 = *(const float4*)(mask + row + s0);
    } else {
      float* mp = (float*)&m4;
      for (int v = 0; v < 4; ++v) if (s0 + v < S_) mp[v] = mask[row + s0 + v];
    }
    const float* mp = (const float*)&m4;
    #pragma unroll
    for (int v = 0; v < 4; ++v) {
      bool nz = (mp[v] != 0.f);
      unsigned long long bal = __ballot(nz);   // wave32: low 32 bits valid
      if (lane == 0) bm[wid][v] = (unsigned)bal;
      if (nz) wv[tid * 4 + v] = mp[v] * mw[row + s0 + v];
    }
    __syncthreads();
    #pragma unroll 1
    for (int w4 = 0; w4 < 4; ++w4) {
      #pragma unroll
      for (int v = 0; v < 4; ++v) {
        unsigned mm = bm[w4][v];
        while (mm) {
          int j = __ffs(mm) - 1;
          mm &= mm - 1;
          int sl = (w4 * 32 + j) * 4 + v;
          acc += xs_t[(size_t)(base + sl) * B_ + tid] * wv[sl];
        }
      }
    }
    __syncthreads();
  }
  gpre_t[(size_t)g * B_ + tid] = acc + mb[g];  // transposed layout [G][B]
}

// ------------------------------------------------------------------
// K4/K7: generic per-row LayerNorm + exact GELU -> f16 activations.
// in element (b, w) at b*rowStride + w*colStride (handles transposed gene buf).
// ------------------------------------------------------------------
__global__ void ln_gelu_rows_k(const float* __restrict__ in, int rowStride, int colStride,
                               const float* __restrict__ lw, const float* __restrict__ lb,
                               _Float16* __restrict__ out, int W) {
  int b = blockIdx.x;
  int tid = threadIdx.x;                 // 256 threads
  __shared__ float red[256];
  float v[16];                           // W <= 4096
  int cnt = 0;
  float sum = 0.f;
  for (int w = tid; w < W; w += 256) {
    float t = in[(size_t)b * rowStride + (size_t)w * colStride];
    v[cnt++] = t;
    sum += t;
  }
  red[tid] = sum; __syncthreads();
  for (int off = 128; off > 0; off >>= 1) {
    if (tid < off) red[tid] += red[tid + off];
    __syncthreads();
  }
  float mean = red[0] / (float)W;
  __syncthreads();
  float ss = 0.f;
  for (int i = 0; i < cnt; ++i) { float d = v[i] - mean; ss += d * d; }
  red[tid] = ss; __syncthreads();
  for (int off = 128; off > 0; off >>= 1) {
    if (tid < off) red[tid] += red[tid + off];
    __syncthreads();
  }
  float inv = rsqrtf(red[0] / (float)W + EPS_);
  cnt = 0;
  for (int w = tid; w < W; w += 256) {
    float t = (v[cnt++] - mean) * inv * lw[w] + lb[w];
    out[(size_t)b * W + w] = (_Float16)gelu_exact(t);
  }
}

// ------------------------------------------------------------------
// K5: f32 -> f16 weight conversion
// ------------------------------------------------------------------
__global__ void f32_to_f16_k(const float* __restrict__ in, _Float16* __restrict__ out, int n) {
  int i = blockIdx.x * 256 + threadIdx.x;
  if (i < n) out[i] = (_Float16)in[i];
}

// ------------------------------------------------------------------
// K6: WMMA GEMM  C[m,n] = sum_k A[m,k] * Bw[n,k] + bias[n]
// A: [M x K] f16 row-major, Bw: [N x K] f16 row-major, C: [M x N] f32.
// One wave (32 threads) per 16x16 tile; K % 32 == 0; no EXEC divergence.
// Lane layouts per CDNA5 ISA 7.12.2:
//   A halves 0..7 = K = hi*8 + (0..7);  halves 8..15 = K = 16 + hi*8 + (0..7)
//   B halves 0..7 = K = hi*8 + (0..7);  halves 8..15 = K = 16 + hi*8 + (0..7), col N = lane%16
//   D c[v] -> row m0 + v + 8*hi, col n0 + lane%16
// ------------------------------------------------------------------
__global__ void wmma_gemm_k(const _Float16* __restrict__ A, const _Float16* __restrict__ Bw,
                            const float* __restrict__ bias, float* __restrict__ C,
                            int M, int N, int K) {
  int n0 = blockIdx.x * 16;
  int m0 = blockIdx.y * 16;
  int lane = threadIdx.x;
  int hi = lane >> 4;
  int lp = lane & 15;
  const _Float16* arow = A + (size_t)(m0 + lp) * K;
  const _Float16* brow = Bw + (size_t)(n0 + lp) * K;
  v8f c = {};
  for (int kk = 0; kk < K; kk += 32) {
    v8h alo = *(const v8h*)(arow + kk + hi * 8);
    v8h ahi = *(const v8h*)(arow + kk + 16 + hi * 8);
    v16h a;
    #pragma unroll
    for (int i = 0; i < 8; ++i) { a[i] = alo[i]; a[i + 8] = ahi[i]; }
    v8h blo = *(const v8h*)(brow + kk + hi * 8);
    v8h bhi = *(const v8h*)(brow + kk + 16 + hi * 8);
    v16h b;
    #pragma unroll
    for (int i = 0; i < 8; ++i) { b[i] = blo[i]; b[i + 8] = bhi[i]; }
    c = __builtin_amdgcn_wmma_f32_16x16x32_f16(false, a, false, b, (short)0, c, false, false);
  }
  float bn = bias[n0 + lp];
  #pragma unroll
  for (int v = 0; v < 8; ++v) {
    int m = m0 + v + 8 * hi;
    C[(size_t)m * N + (n0 + lp)] = c[v] + bn;
  }
}

// ------------------------------------------------------------------
// K8: final projection H2 -> 1
// ------------------------------------------------------------------
__global__ void out_head_k(const _Float16* __restrict__ act, const float* __restrict__ ow,
                           const float* __restrict__ ob, float* __restrict__ y) {
  int b = threadIdx.x;
  if (b >= B_) return;
  float acc = ob[0];
  for (int h = 0; h < H2_; ++h) acc += (float)act[b * H2_ + h] * ow[h];
  y[b] = acc;
}

extern "C" void kernel_launch(void* const* d_in, const int* in_sizes, int n_in,
                              void* d_out, int out_size, void* d_ws, size_t ws_size,
                              hipStream_t stream) {
  const float* x       = (const float*)d_in[0];
  const int*   iidx    = (const int*)d_in[1];
  const float* mask    = (const float*)d_in[2];
  const float* emb     = (const float*)d_in[3];
  const float* proj_w  = (const float*)d_in[4];
  const float* proj_b  = (const float*)d_in[5];
  const float* ln_i_w  = (const float*)d_in[6];
  const float* ln_i_b  = (const float*)d_in[7];
  const float* scale_w = (const float*)d_in[8];
  const float* scale_b = (const float*)d_in[9];
  const float* bias_w  = (const float*)d_in[10];
  const float* bias_b  = (const float*)d_in[11];
  const float* mw      = (const float*)d_in[12];
  const float* mb      = (const float*)d_in[13];
  const float* ln1_w   = (const float*)d_in[14];
  const float* ln1_b   = (const float*)d_in[15];
  const float* fc1_w   = (const float*)d_in[16];
  const float* fc1_b   = (const float*)d_in[17];
  const float* lnA_w   = (const float*)d_in[18];
  const float* lnA_b   = (const float*)d_in[19];
  const float* fc2_w   = (const float*)d_in[20];
  const float* fc2_b   = (const float*)d_in[21];
  const float* lnB_w   = (const float*)d_in[22];
  const float* lnB_b   = (const float*)d_in[23];
  const float* out_w   = (const float*)d_in[24];
  const float* out_b   = (const float*)d_in[25];
  float* y = (float*)d_out;

  // workspace carve-up (256B aligned)
  char* ws = (char*)d_ws;
  size_t off = 0;
  auto take = [&](size_t bytes) -> size_t {
    size_t o = off;
    off += (bytes + 255) & ~(size_t)255;
    return o;
  };
  float*    tab     = (float*)   (ws + take(8 * sizeof(float)));
  float*    xs_t    = (float*)   (ws + take((size_t)S_ * B_ * 4));
  float*    gpre_t  = (float*)   (ws + take((size_t)G_ * B_ * 4));
  _Float16* gact    = (_Float16*)(ws + take((size_t)B_ * G_ * 2));
  _Float16* fc1w_h  = (_Float16*)(ws + take((size_t)H1_ * G_ * 2));
  float*    y1pre   = (float*)   (ws + take((size_t)B_ * H1_ * 4));
  _Float16* act2    = (_Float16*)(ws + take((size_t)B_ * H1_ * 2));
  _Float16* fc2w_h  = (_Float16*)(ws + take((size_t)H2_ * H1_ * 2));
  float*    y2pre   = (float*)   (ws + take((size_t)B_ * H2_ * 4));
  _Float16* act3    = (_Float16*)(ws + take((size_t)B_ * H2_ * 2));

  // 1) 4-entry scale/bias tables
  impact_tables_k<<<1, 32, 0, stream>>>(emb, proj_w, proj_b, ln_i_w, ln_i_b,
                                        scale_w, scale_b, bias_w, bias_b, tab);
  // 2) gated attention + transpose to xs_t[S][B]
  attn_transpose_k<<<dim3(S_ / 32, B_ / 32), dim3(32, 32), 0, stream>>>(x, iidx, tab, xs_t);
  // 3) sparse masked linear -> gpre_t[G][B]
  gene_accum_k<<<G_, 128, 0, stream>>>(mask, mw, xs_t, mb, gpre_t);
  // 4) LN over genes (transposed input) + gelu -> gact f16 [B][G]
  ln_gelu_rows_k<<<B_, 256, 0, stream>>>(gpre_t, 1, B_, ln1_w, ln1_b, gact, G_);
  // 5) weight conversions
  f32_to_f16_k<<<(H1_ * G_ + 255) / 256, 256, 0, stream>>>(fc1_w, fc1w_h, H1_ * G_);
  f32_to_f16_k<<<(H2_ * H1_ + 255) / 256, 256, 0, stream>>>(fc2_w, fc2w_h, H2_ * H1_);
  // 6) fc1 via WMMA (K=4000), bias fused
  wmma_gemm_k<<<dim3(H1_ / 16, B_ / 16), 32, 0, stream>>>(gact, fc1w_h, fc1_b, y1pre,
                                                          B_, H1_, G_);
  // 7) LN(A) + gelu -> act2 f16
  ln_gelu_rows_k<<<B_, 256, 0, stream>>>(y1pre, H1_, 1, lnA_w, lnA_b, act2, H1_);
  // 8) fc2 via WMMA (K=1024)
  wmma_gemm_k<<<dim3(H2_ / 16, B_ / 16), 32, 0, stream>>>(act2, fc2w_h, fc2_b, y2pre,
                                                          B_, H2_, H1_);
  // 9) LN(B) + gelu -> act3 f16
  ln_gelu_rows_k<<<B_, 256, 0, stream>>>(y2pre, H2_, 1, lnB_w, lnB_b, act3, H2_);
  // 10) output head
  out_head_k<<<1, B_, 0, stream>>>(act3, out_w, out_b, y);
}
